// Pair2PairBlock_80711025426886
// MI455X (gfx1250) — compile-verified, matrix operand order checked by reading, not run on hardware
//
#include <hip/hip_runtime.h>
#include <hip/hip_bf16.h>

// ---------------------------------------------------------------------------
// Evoformer pair block for MI455X (gfx1250, wave32, WMMA + Tensor Data Mover).
// All matmuls run through v_wmma_f32_16x16x32_bf16 (bf16 in, f32 accumulate);
// GEMM tiles are staged into LDS by the TDM (tensor_load_to_lds) and gated by
// s_wait_tensorcnt, freeing VALU/VMEM in the main loop.
// ---------------------------------------------------------------------------

#define LL   192
#define CC   256
#define GG   8
#define PPG  32          // channels per attention head
#define FFD  1024
#define NPOS (LL*LL)     // 36864
#define NBUF (NPOS*CC)   // 9437184 floats per [L,L,C] buffer
#define EPSLN 1e-5f

typedef unsigned int       u32;
typedef unsigned long long u64;
typedef __attribute__((ext_vector_type(16))) __bf16 v16bf;
typedef __attribute__((ext_vector_type(8)))  float  v8f;
typedef __attribute__((ext_vector_type(4)))  u32    v4u;
typedef __attribute__((ext_vector_type(8)))  int    v8i;
typedef __attribute__((ext_vector_type(4)))  int    v4i;

// K-index mapping for the CDNA5 16-bit A/B fragment layout (16x32 / 32x16):
// lanes 0-15 hold K∈{0..7,16..23}, lanes 16-31 hold K∈{8..15,24..31}.
__device__ __forceinline__ int kmap(int e, int h) {
  return (e < 8) ? (e + 8 * h) : (e + 8 + 8 * h);
}

// A fragment (16x32) from bf16 LDS: element (m,k) = s[(row0+m)*ld + k0 + k]
__device__ __forceinline__ v16bf frag_a_bf16(const __bf16* s, int row0, int k0, int ld) {
  int lane = threadIdx.x & 31;
  int m = lane & 15, h = lane >> 4;
  v16bf f;
#pragma unroll
  for (int e = 0; e < 16; ++e) f[e] = s[(row0 + m) * ld + k0 + kmap(e, h)];
  return f;
}

// A fragment from an f32 LDS tile (converted while packing)
__device__ __forceinline__ v16bf frag_a_f32(const float* s, int row0, int k0, int ld) {
  int lane = threadIdx.x & 31;
  int m = lane & 15, h = lane >> 4;
  v16bf f;
#pragma unroll
  for (int e = 0; e < 16; ++e) f[e] = (__bf16)s[(row0 + m) * ld + k0 + kmap(e, h)];
  return f;
}

// B fragment (32x16) from bf16 LDS: element (k,n) = s[(k0+k)*ld + col0+n]  ([K][N])
__device__ __forceinline__ v16bf frag_b_kn(const __bf16* s, int k0, int col0, int ld) {
  int lane = threadIdx.x & 31;
  int n = lane & 15, h = lane >> 4;
  v16bf f;
#pragma unroll
  for (int e = 0; e < 16; ++e) f[e] = s[(k0 + kmap(e, h)) * ld + col0 + n];
  return f;
}

// B fragment (32x16) from f32 LDS: element (k,n) = s[(k0+k)*ld + col0+n]
__device__ __forceinline__ v16bf frag_b_kn_f32(const float* s, int k0, int col0, int ld) {
  int lane = threadIdx.x & 31;
  int n = lane & 15, h = lane >> 4;
  v16bf f;
#pragma unroll
  for (int e = 0; e < 16; ++e) f[e] = (__bf16)s[(k0 + kmap(e, h)) * ld + col0 + n];
  return f;
}

// B fragment (32x16): element (k,n) = s[(col0+n)*ld + k0 + k]   (source [N][K])
__device__ __forceinline__ v16bf frag_b_nk(const __bf16* s, int k0, int col0, int ld) {
  int lane = threadIdx.x & 31;
  int n = lane & 15, h = lane >> 4;
  v16bf f;
#pragma unroll
  for (int e = 0; e < 16; ++e) f[e] = s[(col0 + n) * ld + k0 + kmap(e, h)];
  return f;
}

__device__ __forceinline__ v8f wmma_bf16(v16bf a, v16bf b, v8f c) {
  return __builtin_amdgcn_wmma_f32_16x16x32_bf16(false, a, false, b, (short)0, c,
                                                 false, false);
}

__device__ __forceinline__ float sigmoidf_(float x) { return 1.0f / (1.0f + __expf(-x)); }

// ---------------------------------------------------------------------------
// Tensor Data Mover: async 2D f32 tile load Global -> LDS.
// Builds D# group0 (CDNA5 ISA) and group1 (§8.4); groups 2/3 and the
// trailing group are zero (<=2D tensor). OOB reads return zero -> free edges.
//   lds_off          : byte offset of destination in LDS
//   gaddr            : global address of tile origin
//   tensor_w/tensor_h: remaining valid elements / rows from the origin
//   tile_w/tile_h    : tile width (elements) / height (rows)
//   row_stride       : elements between consecutive rows
// ---------------------------------------------------------------------------
__device__ __forceinline__ void tdm_load_2d_f32(u32 lds_off, const float* gaddr,
                                                u32 tensor_w, u32 tensor_h,
                                                u32 tile_w, u32 tile_h,
                                                u32 row_stride) {
  u64 ga = (u64)(size_t)gaddr;
  v4u g0;
  g0[0] = 1u;                                            // count=1 (valid user D#)
  g0[1] = lds_off;                                       // lds_addr (bytes)
  g0[2] = (u32)ga;                                       // global_addr[31:0]
  g0[3] = (u32)((ga >> 32) & 0x1FFFFFFu) | (2u << 30);   // global_addr[56:32]|type=2
  v8i g1;
  g1[0] = (int)(2u << 16);                               // data_size=2 (4 bytes)
  g1[1] = (int)((tensor_w & 0xFFFFu) << 16);             // tensor_dim0[15:0]
  g1[2] = (int)(((tensor_w >> 16) & 0xFFFFu) |           // tensor_dim0[31:16]
                ((tensor_h & 0xFFFFu) << 16));           // tensor_dim1[15:0]
  g1[3] = (int)(((tensor_h >> 16) & 0xFFFFu) |           // tensor_dim1[31:16]
                ((tile_w & 0xFFFFu) << 16));             // tile_dim0
  g1[4] = (int)(tile_h & 0xFFFFu);                       // tile_dim1 (tile_dim2=0)
  g1[5] = (int)row_stride;                               // tensor_dim0_stride[31:0]
  g1[6] = 0;                                             // stride hi / dim1_stride lo
  g1[7] = 0;                                             // dim1_stride hi
  v4i z4 = {0, 0, 0, 0};
  v8i z8 = {0, 0, 0, 0, 0, 0, 0, 0};
  // amdgpu-toolchain (clang-23) 6-arg form: (g0, g1, g2, g3, g4, cpol)
  __builtin_amdgcn_tensor_load_to_lds(g0, g1, z4, z4, z8, 0);
}

// ---------------------------------------------------------------------------
// Generic GEMM: Y[M,N] = epi(A[M,K] @ W[K,N] + bias[N])
// EPI: 0 none | 1 gate: sigmoid(extra)*() | 2 relu | 3 add: extra + ()
// Block: 256 threads (8 waves), tile 128x64, K-step 32; each wave owns 32x32.
// A/W tiles are DMA'd into LDS (f32) by wave 0 via the TDM; completion is
// enforced with s_wait_tensorcnt + workgroup barrier.
// ---------------------------------------------------------------------------
template <int EPI>
__global__ __launch_bounds__(256) void gemm_kernel(
    const float* __restrict__ A, const float* __restrict__ W,
    const float* __restrict__ bias, const float* __restrict__ extra,
    float* __restrict__ Y, int M, int N, int K) {
  __shared__ float Asf[128 * 32];  // [m][k]
  __shared__ float Wsf[32 * 64];   // [k][n]
  const int m0 = blockIdx.y * 128;
  const int n0 = blockIdx.x * 64;
  const int tid = threadIdx.x;
  const int wave = tid >> 5;
  const int wm = wave >> 1, wn = wave & 1;
  const u32 ldsA = (u32)(size_t)Asf;
  const u32 ldsW = (u32)(size_t)Wsf;
  v8f acc[2][2] = {};
  for (int k0 = 0; k0 < K; k0 += 32) {
    __syncthreads();  // previous iteration's fragments consumed
    if (wave == 0) {
      // A tile: 128 rows x 32 f32, row stride K
      tdm_load_2d_f32(ldsA, A + (size_t)m0 * K + k0,
                      (u32)(K - k0), (u32)(M - m0), 32u, 128u, (u32)K);
      // W tile: 32 rows x 64 f32, row stride N (OOB cols of Wb(N=8) read 0)
      tdm_load_2d_f32(ldsW, W + (size_t)k0 * N + n0,
                      (u32)(N - n0), (u32)(K - k0), 64u, 32u, (u32)N);
      __builtin_amdgcn_s_wait_tensorcnt(0);
    }
    __syncthreads();  // publish tiles to all waves
    v16bf a0 = frag_a_f32(Asf, wm * 32, 0, 32);
    v16bf a1 = frag_a_f32(Asf, wm * 32 + 16, 0, 32);
    v16bf b0 = frag_b_kn_f32(Wsf, 0, wn * 32, 64);
    v16bf b1 = frag_b_kn_f32(Wsf, 0, wn * 32 + 16, 64);
    acc[0][0] = wmma_bf16(a0, b0, acc[0][0]);
    acc[0][1] = wmma_bf16(a0, b1, acc[0][1]);
    acc[1][0] = wmma_bf16(a1, b0, acc[1][0]);
    acc[1][1] = wmma_bf16(a1, b1, acc[1][1]);
  }
  const int lane = tid & 31;
  const int nn = lane & 15, h = lane >> 4;
#pragma unroll
  for (int ti = 0; ti < 2; ++ti)
#pragma unroll
    for (int tj = 0; tj < 2; ++tj)
#pragma unroll
      for (int d = 0; d < 8; ++d) {
        int gm = m0 + wm * 32 + ti * 16 + d + 8 * h;
        int gn = n0 + wn * 32 + tj * 16 + nn;
        if (gm < M && gn < N) {
          float v = acc[ti][tj][d] + (bias ? bias[gn] : 0.0f);
          size_t idx = (size_t)gm * N + gn;
          if (EPI == 1)      v = sigmoidf_(extra[idx]) * v;
          else if (EPI == 2) v = fmaxf(v, 0.0f);
          else if (EPI == 3) v = extra[idx] + v;
          Y[idx] = v;
        }
      }
}

// ---------------------------------------------------------------------------
// Triangle einsum: 256 batched per-channel GEMMs over the L(=192) axis.
//  OUTG: O[i,j,c] = sum_k L[i,k,c]*R[j,k,c]   ('nikc,njkc->nijc')
// !OUTG: O[i,j,c] = sum_k L[k,i,c]*R[k,j,c]   ('nkic,nkjc->nijc')
// Gather is stride-C (one element per TDM "line"), so stage with plain loads.
// ---------------------------------------------------------------------------
template <bool OUTG>
__global__ __launch_bounds__(256) void tri_gemm_kernel(
    const float* __restrict__ Lp, const float* __restrict__ Rp,
    float* __restrict__ O) {
  __shared__ __bf16 As[64 * 32];  // [i][k]
  __shared__ __bf16 Bs[32 * 64];  // [k][j]
  const int c = blockIdx.x;
  const int i0 = blockIdx.y * 64;
  const int j0 = blockIdx.z * 64;
  const int tid = threadIdx.x;
  const int wave = tid >> 5;
  const int wm = wave >> 1, wn = wave & 1;
  v8f acc[2] = {};
  for (int k0 = 0; k0 < LL; k0 += 32) {
    __syncthreads();
#pragma unroll
    for (int e = tid; e < 64 * 32; e += 256) {
      int r = e >> 5, k = e & 31;
      size_t src = OUTG ? ((size_t)((i0 + r) * LL + (k0 + k)) * CC + c)
                        : ((size_t)((k0 + k) * LL + (i0 + r)) * CC + c);
      As[e] = (__bf16)Lp[src];
    }
#pragma unroll
    for (int e = tid; e < 32 * 64; e += 256) {
      int k = e >> 6, j = e & 63;
      size_t src = OUTG ? ((size_t)((j0 + j) * LL + (k0 + k)) * CC + c)
                        : ((size_t)((k0 + k) * LL + (j0 + j)) * CC + c);
      Bs[e] = (__bf16)Rp[src];
    }
    __syncthreads();
    v16bf a = frag_a_bf16(As, wm * 16, 0, 32);
#pragma unroll
    for (int tj = 0; tj < 2; ++tj) {
      v16bf b = frag_b_kn(Bs, 0, wn * 32 + tj * 16, 64);
      acc[tj] = wmma_bf16(a, b, acc[tj]);
    }
  }
  const int lane = tid & 31;
  const int nn = lane & 15, h = lane >> 4;
#pragma unroll
  for (int tj = 0; tj < 2; ++tj)
#pragma unroll
    for (int d = 0; d < 8; ++d) {
      int gi = i0 + wm * 16 + d + 8 * h;
      int gj = j0 + wn * 32 + tj * 16 + nn;
      O[(size_t)(gi * LL + gj) * CC + c] = acc[tj][d];
    }
}

// ---------------------------------------------------------------------------
// LayerNorm over C=256: one wave32 per row (8 channels/lane), 8 rows/block.
// ---------------------------------------------------------------------------
__global__ __launch_bounds__(256) void layernorm_kernel(
    const float* __restrict__ X, const float* __restrict__ g,
    const float* __restrict__ b, float* __restrict__ Y, int M) {
  const int wave = threadIdx.x >> 5;
  const int lane = threadIdx.x & 31;
  const int row = blockIdx.x * 8 + wave;
  if (row >= M) return;
  const float* xr = X + (size_t)row * CC;
  float v[8];
  float s = 0.f;
#pragma unroll
  for (int t = 0; t < 8; ++t) { v[t] = xr[lane + 32 * t]; s += v[t]; }
#pragma unroll
  for (int m = 16; m >= 1; m >>= 1) s += __shfl_xor(s, m, 32);
  float mean = s * (1.0f / CC);
  float q = 0.f;
#pragma unroll
  for (int t = 0; t < 8; ++t) { float d = v[t] - mean; q += d * d; }
#pragma unroll
  for (int m = 16; m >= 1; m >>= 1) q += __shfl_xor(q, m, 32);
  float inv = rsqrtf(q * (1.0f / CC) + EPSLN);
  float* yr = Y + (size_t)row * CC;
#pragma unroll
  for (int t = 0; t < 8; ++t) {
    int ch = lane + 32 * t;
    yr[ch] = (v[t] - mean) * inv * g[ch] + b[ch];
  }
}

// ---------------------------------------------------------------------------
// Elementwise helpers
// ---------------------------------------------------------------------------
__global__ void add_kernel(float* __restrict__ X, const float* __restrict__ T, int n) {
  for (int i = blockIdx.x * blockDim.x + threadIdx.x; i < n; i += gridDim.x * blockDim.x)
    X[i] += T[i];
}
__global__ void add_t_kernel(float* __restrict__ X, const float* __restrict__ T) {
  for (int idx = blockIdx.x * blockDim.x + threadIdx.x; idx < NBUF;
       idx += gridDim.x * blockDim.x) {
    int c = idx & (CC - 1);
    int pos = idx >> 8;
    int j = pos % LL, i = pos / LL;
    X[idx] += T[(size_t)(j * LL + i) * CC + c];
  }
}
__global__ void transpose_kernel(const float* __restrict__ X, float* __restrict__ T) {
  for (int idx = blockIdx.x * blockDim.x + threadIdx.x; idx < NBUF;
       idx += gridDim.x * blockDim.x) {
    int c = idx & (CC - 1);
    int pos = idx >> 8;
    int j = pos % LL, i = pos / LL;
    T[(size_t)(j * LL + i) * CC + c] = X[idx];
  }
}

// ---------------------------------------------------------------------------
// Fused triangle attention: one workgroup per (head g, row i).
// S[j,k] = q[i,j,:]·k[i,k,:]/sqrt(ppg) + bias[j,k,g]; softmax over j (axis 2,
// faithful to the reference); O[j,c] = sigmoid(gate)*sum_k P[j,k]*v[i,k,c].
// Full 192x192 f32 logit matrix lives in LDS (pitch 196 to dodge bank 0).
// ---------------------------------------------------------------------------
#define SPITCH 196
#define ATT_LDS_BYTES (LL * SPITCH * 4 + 3 * LL * PPG * 2)  // 187392 B < 320 KB

__global__ __launch_bounds__(256) void attn_kernel(
    const float* __restrict__ Q, const float* __restrict__ Kx,
    const float* __restrict__ V, const float* __restrict__ GL,
    const float* __restrict__ Bb, float* __restrict__ O) {
  extern __shared__ char smem[];
  float* S = (float*)smem;
  __bf16* qs = (__bf16*)(smem + LL * SPITCH * 4);
  __bf16* ks = qs + LL * PPG;
  __bf16* vs = ks + LL * PPG;
  const int g = blockIdx.x;
  const int i = blockIdx.y;
  const int tid = threadIdx.x;
  const int wave = tid >> 5, lane = tid & 31;
  const size_t base = (size_t)i * LL * CC + g * PPG;
  const float scale = 0.17677669529663689f;  // 1/sqrt(32)
  for (int e = tid; e < LL * PPG; e += 256) {
    int j = e >> 5, c = e & 31;
    size_t src = base + (size_t)j * CC + c;
    qs[e] = (__bf16)(Q[src] * scale);
    ks[e] = (__bf16)Kx[src];
    vs[e] = (__bf16)V[src];
  }
  __syncthreads();
  const int nn = lane & 15, h = lane >> 4;
  // ---- logits: 12x12 tiles of 16x16, one 16x16x32 WMMA each ----
  for (int t = wave; t < 144; t += 8) {
    int ti = t / 12, tj = t % 12;
    v16bf a = frag_a_bf16(qs, ti * 16, 0, PPG);
    v16bf b = frag_b_nk(ks, 0, tj * 16, PPG);
    v8f acc = {};
    acc = wmma_bf16(a, b, acc);
#pragma unroll
    for (int d = 0; d < 8; ++d) {
      int jr = ti * 16 + d + 8 * h;
      int kc = tj * 16 + nn;
      S[jr * SPITCH + kc] = acc[d] + Bb[(size_t)(jr * LL + kc) * GG + g];
    }
  }
  __syncthreads();
  // ---- softmax over j (rows) for each column k ----
  if (tid < LL) {
    int kc = tid;
    float mx = -1e30f;
    for (int j = 0; j < LL; ++j) mx = fmaxf(mx, S[j * SPITCH + kc]);
    float sum = 0.f;
    for (int j = 0; j < LL; ++j) {
      float e = __expf(S[j * SPITCH + kc] - mx);
      S[j * SPITCH + kc] = e;
      sum += e;
    }
    float inv = 1.0f / sum;
    for (int j = 0; j < LL; ++j) S[j * SPITCH + kc] *= inv;
  }
  __syncthreads();
  // ---- O = P @ V (192x32x192), gate fused into the store ----
  for (int t = wave; t < 24; t += 8) {
    int ti = t >> 1, tj = t & 1;
    v8f acc = {};
    for (int kk0 = 0; kk0 < LL; kk0 += 32) {
      v16bf a = frag_a_f32(S, ti * 16, kk0, SPITCH);
      v16bf b = frag_b_kn(vs, kk0, tj * 16, PPG);
      acc = wmma_bf16(a, b, acc);
    }
#pragma unroll
    for (int d = 0; d < 8; ++d) {
      int jr = ti * 16 + d + 8 * h;
      int c = tj * 16 + nn;
      size_t idx = base + (size_t)jr * CC + c;
      O[idx] = sigmoidf_(GL[idx]) * acc[d];
    }
  }
}

// ---------------------------------------------------------------------------
// Host orchestration
// ---------------------------------------------------------------------------
// Input flattening (setup_inputs dict order, nested dicts in insertion order):
enum {
  IN_X = 0,
  IN_N1G, IN_N1B, IN_N2G, IN_N2B, IN_N3G, IN_N3B, IN_N4G, IN_N4B,  // 1..8
  TMO_BASE = 9,   // Wl1,bl1,Wl2,bl2,Wr1,br1,Wr2,br2,ln_g,ln_b,Wg,bg,Wo,bo
  TMI_BASE = 23,
  TAO_BASE = 37,  // Wq,Wk,Wv,Wb,Wg,bg,Wo,bo
  TAI_BASE = 45,
  FF_LNG = 53, FF_LNB, FF_W1, FF_B1, FF_W2, FF_B2
};

extern "C" void kernel_launch(void* const* d_in, const int* in_sizes, int n_in,
                              void* d_out, int out_size, void* d_ws, size_t ws_size,
                              hipStream_t stream) {
  (void)in_sizes; (void)n_in; (void)out_size; (void)ws_size;
  auto P = [&](int i) { return (const float*)d_in[i]; };
  float* out = (float*)d_out;

  // Workspace: 6 full [L,L,C] f32 buffers + [L,L,G] attention-bias buffer.
  float* xb = (float*)d_ws;
  float* t0 = xb + 1 * (size_t)NBUF;
  float* t1 = xb + 2 * (size_t)NBUF;
  float* t2 = xb + 3 * (size_t)NBUF;
  float* t3 = xb + 4 * (size_t)NBUF;
  float* t4 = xb + 5 * (size_t)NBUF;
  float* bb = xb + 6 * (size_t)NBUF;

  (void)hipMemcpyAsync(xb, d_in[IN_X], (size_t)NBUF * sizeof(float),
                       hipMemcpyDeviceToDevice, stream);

  auto gemm = [&](const float* A, const float* W, const float* bias,
                  const float* extra, float* Y, int M, int N, int K, int epi) {
    dim3 grid((N + 63) / 64, (M + 127) / 128);
    dim3 blk(256);
    switch (epi) {
      case 0: gemm_kernel<0><<<grid, blk, 0, stream>>>(A, W, bias, extra, Y, M, N, K); break;
      case 1: gemm_kernel<1><<<grid, blk, 0, stream>>>(A, W, bias, extra, Y, M, N, K); break;
      case 2: gemm_kernel<2><<<grid, blk, 0, stream>>>(A, W, bias, extra, Y, M, N, K); break;
      default: gemm_kernel<3><<<grid, blk, 0, stream>>>(A, W, bias, extra, Y, M, N, K); break;
    }
  };
  auto ln = [&](const float* X, const float* g_, const float* b_, float* Y) {
    layernorm_kernel<<<dim3(NPOS / 8), dim3(256), 0, stream>>>(X, g_, b_, Y, NPOS);
  };
  auto add = [&](float* X, const float* T) {
    add_kernel<<<dim3(2048), dim3(256), 0, stream>>>(X, T, NBUF);
  };

  // ======== triangle multiplication (outgoing, then incoming) ========
  auto trimul = [&](int B, bool outg, const float* png, const float* pnb) {
    ln(xb, png, pnb, t0);
    gemm(t0, P(B + 0), P(B + 1), nullptr, t1, NPOS, CC, CC, 0);  // left gate logits
    gemm(t0, P(B + 2), P(B + 3), t1,      t2, NPOS, CC, CC, 1);  // left  = sig(l1)*l2
    gemm(t0, P(B + 4), P(B + 5), nullptr, t1, NPOS, CC, CC, 0);  // right gate logits
    gemm(t0, P(B + 6), P(B + 7), t1,      t3, NPOS, CC, CC, 1);  // right = sig(r1)*r2
    dim3 tg(CC, LL / 64, LL / 64);
    if (outg) tri_gemm_kernel<true ><<<tg, dim3(256), 0, stream>>>(t2, t3, t1);
    else      tri_gemm_kernel<false><<<tg, dim3(256), 0, stream>>>(t2, t3, t1);
    ln(t1, P(B + 8), P(B + 9), t0);
    gemm(t0, P(B + 10), P(B + 11), nullptr, t2, NPOS, CC, CC, 0);  // out gate logits
    gemm(t0, P(B + 12), P(B + 13), t2,      t1, NPOS, CC, CC, 1);  // sig(g)*(Wo out)
    add(xb, t1);
  };
  trimul(TMO_BASE, true,  P(IN_N1G), P(IN_N1B));
  trimul(TMI_BASE, false, P(IN_N2G), P(IN_N2B));

  // ======== triangle attention (outgoing) ========
  {
    const int B = TAO_BASE;
    ln(xb, P(IN_N3G), P(IN_N3B), t0);
    gemm(t0, P(B + 3), nullptr,  nullptr, bb, NPOS, GG, CC, 0);  // bias = ln @ Wb
    gemm(t0, P(B + 4), P(B + 5), nullptr, t1, NPOS, CC, CC, 0);  // gate logits
    gemm(t0, P(B + 0), nullptr,  nullptr, t2, NPOS, CC, CC, 0);  // q
    gemm(t0, P(B + 1), nullptr,  nullptr, t3, NPOS, CC, CC, 0);  // k
    gemm(t0, P(B + 2), nullptr,  nullptr, t4, NPOS, CC, CC, 0);  // v
    attn_kernel<<<dim3(GG, LL), dim3(256), ATT_LDS_BYTES, stream>>>(t2, t3, t4, t1, bb, t0);
    gemm(t0, P(B + 6), P(B + 7), nullptr, t1, NPOS, CC, CC, 0);  // o @ Wo + bo
    add(xb, t1);
  }

  // ======== triangle attention (incoming: operate on transposed x) ========
  {
    const int B = TAI_BASE;
    transpose_kernel<<<dim3(2048), dim3(256), 0, stream>>>(xb, t0);
    ln(t0, P(IN_N4G), P(IN_N4B), t1);
    gemm(t1, P(B + 3), nullptr,  nullptr, bb, NPOS, GG, CC, 0);
    gemm(t1, P(B + 4), P(B + 5), nullptr, t2, NPOS, CC, CC, 0);  // gate logits
    gemm(t1, P(B + 0), nullptr,  nullptr, t3, NPOS, CC, CC, 0);  // q
    gemm(t1, P(B + 1), nullptr,  nullptr, t4, NPOS, CC, CC, 0);  // k
    gemm(t1, P(B + 2), nullptr,  nullptr, t0, NPOS, CC, CC, 0);  // v (xT dead)
    attn_kernel<<<dim3(GG, LL), dim3(256), ATT_LDS_BYTES, stream>>>(t3, t4, t0, t2, bb, t1);
    gemm(t1, P(B + 6), P(B + 7), nullptr, t2, NPOS, CC, CC, 0);
    add_t_kernel<<<dim3(2048), dim3(256), 0, stream>>>(xb, t2);   // x += oT
  }

  // ======== feed-forward transition (chunked hidden, writes d_out) ========
  {
    ln(xb, P(FF_LNG), P(FF_LNB), t0);
    const int MCH = NPOS / 4;  // 9216 rows/chunk -> hidden chunk fits one buffer
    for (int c = 0; c < 4; ++c) {
      size_t off = (size_t)c * MCH * CC;
      gemm(t0 + off, P(FF_W1), P(FF_B1), nullptr, t1, MCH, FFD, CC, 2);   // relu
      gemm(t1, P(FF_W2), P(FF_B2), xb + off, out + off, MCH, CC, FFD, 3); // x + ...
    }
  }
}
